// ProteinFeatures_24764781429445
// MI455X (gfx1250) — compile-verified, matrix operand order checked by reading, not run on hardware
//
#include <hip/hip_runtime.h>
#include <hip/hip_bf16.h>

#define N_RES    8192
#define KNN      48
#define POS_OH   66
#define POS_DIM  16
#define EDGE_IN  416     // 16 + 5*5*16
#define EDGE_F   128
#define M_ROWS   (N_RES * KNN)        // 393216
#define E_ELEMS  (M_ROWS * EDGE_F)    // 50331648

typedef __attribute__((ext_vector_type(16))) _Float16 v16h;
typedef __attribute__((ext_vector_type(8)))  float    v8f;

union FragAB { v16h v; uint4 u[2]; };

__device__ __forceinline__ void lds_fence() {
  asm volatile("s_wait_dscnt 0" ::: "memory");
}

// ---------------------------------------------------------------------------
// Kernel 1: noisy coords -> backbone frame (N, CA, C, O, CB) + f16 weight prep
// ---------------------------------------------------------------------------
__global__ __launch_bounds__(256)
void prep_kernel(const float* __restrict__ sc, const float* __restrict__ noise,
                 const float* __restrict__ bn,
                 const float* __restrict__ w_e_w, const float* __restrict__ proj_w,
                 float* __restrict__ bb,
                 _Float16* __restrict__ w_e_wh, _Float16* __restrict__ proj_wh)
{
  const int t = blockIdx.x * 256 + threadIdx.x;
  if (t < N_RES) {
    const float s = bn[0];
    const int b0 = t * 37 * 3;
    float n_[3], ca[3], c_[3], o_[3], cb[3];
#pragma unroll
    for (int c = 0; c < 3; ++c) {
      n_[c] = sc[b0 + 0 * 3 + c] + s * noise[b0 + 0 * 3 + c];
      ca[c] = sc[b0 + 1 * 3 + c] + s * noise[b0 + 1 * 3 + c];
      c_[c] = sc[b0 + 2 * 3 + c] + s * noise[b0 + 2 * 3 + c];
      o_[c] = sc[b0 + 4 * 3 + c] + s * noise[b0 + 4 * 3 + c];
    }
    float bx = ca[0] - n_[0], by = ca[1] - n_[1], bz = ca[2] - n_[2];
    float cx = c_[0] - ca[0], cy = c_[1] - ca[1], cz = c_[2] - ca[2];
    float ax = by * cz - bz * cy;
    float ay = bz * cx - bx * cz;
    float az = bx * cy - by * cx;
    cb[0] = -0.58273431f * ax + 0.56802827f * bx - 0.54067466f * cx + ca[0];
    cb[1] = -0.58273431f * ay + 0.56802827f * by - 0.54067466f * cy + ca[1];
    cb[2] = -0.58273431f * az + 0.56802827f * bz - 0.54067466f * cz + ca[2];
    float* o = bb + t * 15;
#pragma unroll
    for (int c = 0; c < 3; ++c) {
      o[0 * 3 + c] = n_[c]; o[1 * 3 + c] = ca[c]; o[2 * 3 + c] = c_[c];
      o[3 * 3 + c] = o_[c]; o[4 * 3 + c] = cb[c];
    }
  } else if (t < N_RES + EDGE_F * EDGE_IN) {
    const int q = t - N_RES;
    w_e_wh[q] = (_Float16)w_e_w[q];
  } else if (t < N_RES + EDGE_F * EDGE_IN + EDGE_F * EDGE_F) {
    const int q = t - (N_RES + EDGE_F * EDGE_IN);
    proj_wh[q] = (_Float16)proj_w[q];
  }
}

// ---------------------------------------------------------------------------
// Kernel 2: top-K=48 nearest CA neighbors; one block (256 thr) per residue.
// Distances cached in LDS; 48 argmin passes with lower-index tie-break.
// ---------------------------------------------------------------------------
__global__ __launch_bounds__(256)
void topk_kernel(const float* __restrict__ bb, const float* __restrict__ mask,
                 int* __restrict__ nidx_ws, int* __restrict__ nidx_out)
{
  __shared__ float dist[N_RES];
  __shared__ float rv[256];
  __shared__ int   ri[256];
  const int i = blockIdx.x;
  const int tid = threadIdx.x;
  const float INF = __builtin_inff();
  const float cx = bb[i * 15 + 3], cy = bb[i * 15 + 4], cz = bb[i * 15 + 5];
  const float mi = mask[i];
  for (int j = tid; j < N_RES; j += 256) {
    float dx = cx - bb[j * 15 + 3];
    float dy = cy - bb[j * 15 + 4];
    float dz = cz - bb[j * 15 + 5];
    float d = __fsqrt_rn(dx * dx + dy * dy + dz * dz + 1e-6f);
    dist[j] = (mi * mask[j] != 0.0f) ? d : INF;
  }
  __syncthreads();
  for (int k = 0; k < KNN; ++k) {
    float bv = INF; int bi = 0x7fffffff;
    for (int j = tid; j < N_RES; j += 256) {
      float v = dist[j];
      if (v < bv) { bv = v; bi = j; }      // ascending j => first min = lowest idx
    }
    rv[tid] = bv; ri[tid] = bi;
    __syncthreads();
    for (int s = 128; s > 0; s >>= 1) {
      if (tid < s) {
        float ov = rv[tid + s]; int oi = ri[tid + s];
        if (ov < rv[tid] || (ov == rv[tid] && oi < ri[tid])) { rv[tid] = ov; ri[tid] = oi; }
      }
      __syncthreads();
    }
    if (tid == 0) {
      nidx_ws[i * KNN + k]  = ri[0];
      nidx_out[i * KNN + k] = ri[0];
      dist[ri[0]] = INF;
    }
    __syncthreads();
  }
}

// ---------------------------------------------------------------------------
// Kernel 3: fused edge features -> WMMA GEMM1 (416x128) -> LayerNorm ->
//           WMMA GEMM2 (128x128) -> output.  One wave per 16-row tile,
//           4 waves / block.  All tile data lives in a per-wave LDS region.
//           B fragments for a K-step are loaded as one batch so the 8 WMMAs
//           issue back-to-back on the matrix pipe.
// ---------------------------------------------------------------------------
__global__ __launch_bounds__(128)
void fused_edge_gemm(const float* __restrict__ bb,
                     const int* __restrict__ nidx,
                     const int* __restrict__ res_idx,
                     const int* __restrict__ chain_idx,
                     const float* __restrict__ w_pos_w,
                     const float* __restrict__ w_pos_b,
                     const _Float16* __restrict__ w_e_wh,
                     const float* __restrict__ ln_g,
                     const float* __restrict__ ln_b,
                     const _Float16* __restrict__ proj_wh,
                     const float* __restrict__ proj_b,
                     float* __restrict__ e_out)
{
  // per-wave region: [0,6656) f16 features; reused as [0,2048) f32 e1,
  // then [4096,6144) f16 normalized activations.
  __shared__ __align__(16) _Float16 shW[4][16 * EDGE_IN];
  __shared__ float shAtoms[4][32];

  const int w    = threadIdx.x >> 5;
  const int lane = threadIdx.x & 31;
  const int half = lane >> 4;
  const int idx  = lane & 15;

  _Float16* feat = shW[w];
  float*    e1   = reinterpret_cast<float*>(shW[w]);
  _Float16* nrm  = shW[w] + 4096;
  float*    atomsW = shAtoms[w];

  const int tile    = blockIdx.x * 4 + w;
  const int rowBase = tile * 16;

  // warm WGP$/L2 with the (shared, re-read) f16 weights: 139264 bytes total
  {
    const char* wbase = (const char*)w_e_wh;
    for (int off = (int)threadIdx.x * 64; off < 139264; off += 128 * 64)
      __builtin_prefetch(wbase + off, 0, 3);
  }

  // ---- Phase 1: build 16 x 416 edge-feature tile (f32 math, f16 store) ----
  for (int m = 0; m < 16; ++m) {
    const int row = rowBase + m;
    const int i = row / KNN;
    const int j = nidx[row];
    if (lane < 15) atomsW[lane] = bb[i * 15 + lane];
    const int l2 = lane - 16;
    if (l2 >= 0 && l2 < 15) atomsW[16 + l2] = bb[j * 15 + l2];
    lds_fence();

    const int off   = res_idx[i] - res_idx[j];
    const int samec = (chain_idx[i] == chain_idx[j]) ? 1 : 0;
    int offc = off + 32;
    offc = offc < 0 ? 0 : (offc > 64 ? 64 : offc);
    const int enc = samec ? offc : 65;

    for (int f = lane; f < EDGE_IN; f += 32) {
      float val;
      if (f < POS_DIM) {
        val = w_pos_w[f * POS_OH + enc] + w_pos_b[f];
      } else {
        const int g    = f - POS_DIM;
        const int a    = g / 80;          // own atom
        const int rem  = g - a * 80;
        const int bAt  = rem >> 4;        // neighbor atom
        const int bin  = rem & 15;
        float dx = atomsW[a * 3 + 0] - atomsW[16 + bAt * 3 + 0];
        float dy = atomsW[a * 3 + 1] - atomsW[16 + bAt * 3 + 1];
        float dz = atomsW[a * 3 + 2] - atomsW[16 + bAt * 3 + 2];
        float d  = __fsqrt_rn(dx * dx + dy * dy + dz * dz + 1e-6f);
        float mu = 2.0f + (20.0f / 15.0f) * (float)bin;
        float t  = (d - mu) * 0.8f;       // 1/sigma, sigma = 1.25
        val = __expf(-t * t);
      }
      feat[m * EDGE_IN + f] = (_Float16)val;
    }
  }
  lds_fence();

  // ---- Phase 2: GEMM1  e1[16x128] = feat[16x416] @ w_e_w.T ----
  const v8f zero8 = {0.f, 0.f, 0.f, 0.f, 0.f, 0.f, 0.f, 0.f};
  v8f acc[8];
#pragma unroll
  for (int t = 0; t < 8; ++t) acc[t] = zero8;

  for (int ks = 0; ks < 13; ++ks) {
    const int ko = ks * 32 + 8 * half;    // lane's first K element (16B aligned)
    FragAB a;
    a.u[0] = *reinterpret_cast<const uint4*>(feat + idx * EDGE_IN + ko);
    a.u[1] = *reinterpret_cast<const uint4*>(feat + idx * EDGE_IN + ko + 16);
    FragAB bf[8];
#pragma unroll
    for (int t = 0; t < 8; ++t) {
      const _Float16* bp = w_e_wh + (t * 16 + idx) * EDGE_IN + ko;
      bf[t].u[0] = *reinterpret_cast<const uint4*>(bp);
      bf[t].u[1] = *reinterpret_cast<const uint4*>(bp + 16);
    }
#pragma unroll
    for (int t = 0; t < 8; ++t)
      acc[t] = __builtin_amdgcn_wmma_f32_16x16x32_f16(
          false, a.v, false, bf[t].v, (short)0, acc[t], false, false);
  }

  // spill e1 tile to LDS (overwrites feature region; LDS ops are in order)
  lds_fence();
#pragma unroll
  for (int t = 0; t < 8; ++t)
#pragma unroll
    for (int v = 0; v < 8; ++v)
      e1[(v + 8 * half) * EDGE_F + t * 16 + idx] = acc[t][v];
  lds_fence();

  // ---- Phase 3: LayerNorm(128) per row; lanes 0..15 each own one row ----
  if (lane < 16) {
    const int r = lane;
    float s = 0.f, s2 = 0.f;
    for (int n = 0; n < EDGE_F; ++n) {
      float x = e1[r * EDGE_F + n];
      s += x; s2 += x * x;
    }
    const float mu   = s * (1.0f / 128.0f);
    const float var  = s2 * (1.0f / 128.0f) - mu * mu;
    const float rstd = rsqrtf(var + 1e-5f);
    for (int n = 0; n < EDGE_F; ++n) {
      float x = (e1[r * EDGE_F + n] - mu) * rstd * ln_g[n] + ln_b[n];
      nrm[r * EDGE_F + n] = (_Float16)x;
    }
  }
  lds_fence();

  // ---- Phase 4: GEMM2  out[16x128] = nrm[16x128] @ proj_w.T + proj_b ----
  v8f acc2[8];
#pragma unroll
  for (int t = 0; t < 8; ++t) acc2[t] = zero8;

  for (int ks = 0; ks < 4; ++ks) {
    const int ko = ks * 32 + 8 * half;
    FragAB a;
    a.u[0] = *reinterpret_cast<const uint4*>(nrm + idx * EDGE_F + ko);
    a.u[1] = *reinterpret_cast<const uint4*>(nrm + idx * EDGE_F + ko + 16);
    FragAB bf[8];
#pragma unroll
    for (int t = 0; t < 8; ++t) {
      const _Float16* bp = proj_wh + (t * 16 + idx) * EDGE_F + ko;
      bf[t].u[0] = *reinterpret_cast<const uint4*>(bp);
      bf[t].u[1] = *reinterpret_cast<const uint4*>(bp + 16);
    }
#pragma unroll
    for (int t = 0; t < 8; ++t)
      acc2[t] = __builtin_amdgcn_wmma_f32_16x16x32_f16(
          false, a.v, false, bf[t].v, (short)0, acc2[t], false, false);
  }

#pragma unroll
  for (int t = 0; t < 8; ++t) {
    const float pb = proj_b[t * 16 + idx];
#pragma unroll
    for (int v = 0; v < 8; ++v) {
      const int rowg = rowBase + v + 8 * half;
      e_out[rowg * EDGE_F + t * 16 + idx] = acc2[t][v] + pb;
    }
  }
}

// ---------------------------------------------------------------------------
extern "C" void kernel_launch(void* const* d_in, const int* in_sizes, int n_in,
                              void* d_out, int out_size, void* d_ws, size_t ws_size,
                              hipStream_t stream) {
  const float* sc        = (const float*)d_in[0];
  const float* noise     = (const float*)d_in[1];
  const float* bn        = (const float*)d_in[2];
  const float* mask      = (const float*)d_in[3];
  const int*   res_idx   = (const int*)d_in[4];
  const int*   chain_idx = (const int*)d_in[5];
  const float* w_pos_w   = (const float*)d_in[6];
  const float* w_pos_b   = (const float*)d_in[7];
  const float* w_e_w     = (const float*)d_in[8];
  const float* ln_g      = (const float*)d_in[9];
  const float* ln_b      = (const float*)d_in[10];
  const float* proj_w    = (const float*)d_in[11];
  const float* proj_b    = (const float*)d_in[12];

  // workspace layout (float units): bb | w_e_w(f16) | proj_w(f16) | nidx(int)
  float*    ws      = (float*)d_ws;
  float*    bb      = ws;                                       // 8192*15
  _Float16* w_e_wh  = (_Float16*)(ws + N_RES * 15);             // 53248 halves
  _Float16* proj_wh = (_Float16*)(ws + N_RES * 15 + (EDGE_F * EDGE_IN) / 2);
  int*      nidx_ws = (int*)(ws + N_RES * 15 + (EDGE_F * EDGE_IN) / 2
                                 + (EDGE_F * EDGE_F) / 2);      // 393216 ints

  float* e_out    = (float*)d_out;
  int*   nidx_out = (int*)(e_out + E_ELEMS);   // tuple output #2, int32 bits

  const int prep_items = N_RES + EDGE_F * EDGE_IN + EDGE_F * EDGE_F;
  prep_kernel<<<(prep_items + 255) / 256, 256, 0, stream>>>(
      sc, noise, bn, w_e_w, proj_w, bb, w_e_wh, proj_wh);

  topk_kernel<<<N_RES, 256, 0, stream>>>(bb, mask, nidx_ws, nidx_out);

  // 393216 rows / 16 per wave / 4 waves per block = 6144 blocks
  fused_edge_gemm<<<M_ROWS / 64, 128, 0, stream>>>(
      bb, nidx_ws, res_idx, chain_idx, w_pos_w, w_pos_b, w_e_wh,
      ln_g, ln_b, proj_wh, proj_b, e_out);
}
